// PVConv2d_54683523613037
// MI455X (gfx1250) — matrix operand order repaired and checked by reading, not compiled, not run on hardware
//
#include <hip/hip_runtime.h>

#define KH 3
#define KW 3
#define CIN 32
#define COUT 32
#define HH 256
#define WW 256
#define BB 4
#define HP 258
#define WP 258
#define SQC 0.1f
#define EPSN 1e-7f

typedef __attribute__((ext_vector_type(16))) __bf16 bf16x16;
typedef __attribute__((ext_vector_type(8)))  __bf16 bf16x8;
typedef __attribute__((ext_vector_type(8)))  float  f32x8;

union AFrag { bf16x16 v; bf16x8 h[2]; };

__device__ __forceinline__ __bf16 f2bf(float f) {
    unsigned u = __float_as_uint(f);
    unsigned r = (u + 0x7FFFu + ((u >> 16) & 1u)) >> 16;
    unsigned short s = (unsigned short)r;
    return __builtin_bit_cast(__bf16, s);
}

// ---------------------------------------------------------------------------
// Kernel 1: U = log0(x) into padded bf16 NHWC buffer (halo written as zeros).
// x: [B, 32, 256, 256] f32 NCHW  ->  U: [B, 258, 258, 32] bf16
// ---------------------------------------------------------------------------
__global__ __launch_bounds__(256) void log0_kernel(const float* __restrict__ x,
                                                   __bf16* __restrict__ U) {
    const int PP = HP * WP;
    int idx = blockIdx.x * blockDim.x + threadIdx.x;
    if (idx >= BB * PP) return;
    int b  = idx / PP;
    int r  = idx - b * PP;
    int hp = r / WP;
    int wp = r - hp * WP;

    __bf16* dst = U + (size_t)idx * CIN;
    float vals[CIN];

    if (hp >= 1 && hp <= HH && wp >= 1 && wp <= WW) {
        const float* src = x + (size_t)b * CIN * HH * WW
                             + (size_t)(hp - 1) * WW + (wp - 1);
        float ss = 0.f;
        #pragma unroll
        for (int c = 0; c < CIN; ++c) {
            float v = src[(size_t)c * (HH * WW)];
            vals[c] = v;
            ss += v * v;
        }
        float n = fmaxf(sqrtf(ss), EPSN);
        float t = fminf(SQC * n, 1.0f - 1e-6f);
        float at = 0.5f * __logf((1.f + t) / (1.f - t));   // artanh(t)
        float f = at / (SQC * n);
        #pragma unroll
        for (int c = 0; c < CIN; ++c) vals[c] *= f;
    } else {
        #pragma unroll
        for (int c = 0; c < CIN; ++c) vals[c] = 0.f;
    }

    #pragma unroll
    for (int g = 0; g < 4; ++g) {
        bf16x8 pk;
        #pragma unroll
        for (int e = 0; e < 8; ++e) pk[e] = f2bf(vals[g * 8 + e]);
        *(bf16x8*)(dst + g * 8) = pk;
    }
}

// ---------------------------------------------------------------------------
// Kernel 2: pre-swizzle Wk[9][Cout][Cin] f32 into per-lane bf16 B-fragments.
// Layout: Wb[tap][half][lane][e]; lane holds column N = half*16 + (lane&15),
// elements e: K = 8*(lane>>4) + e for e<8, 16 + 8*(lane>>4) + (e-8) for e>=8.
// ---------------------------------------------------------------------------
__global__ void wswz_kernel(const float* __restrict__ Wk, __bf16* __restrict__ Wb) {
    int idx = blockIdx.x * blockDim.x + threadIdx.x;   // 9*2*32*16 = 9216
    if (idx >= KH * KW * 2 * 32 * 16) return;
    int e    = idx & 15;
    int lane = (idx >> 4) & 31;
    int half = (idx >> 9) & 1;
    int k    = idx >> 10;
    int N = half * 16 + (lane & 15);
    int K = (e & 7) + ((lane >> 4) << 3) + ((e >> 3) << 4);
    Wb[idx] = f2bf(Wk[(k * COUT + N) * CIN + K]);
}

// ---------------------------------------------------------------------------
// Kernel 3: 3x3 conv via v_wmma_f32_16x16x32_bf16 (K=32 == Cin),
// + bias + relu + exp0 + NHWC->NCHW transpose store.
// One wave computes a tile of 16 consecutive pixels x 32 Cout.
// ---------------------------------------------------------------------------
__global__ __launch_bounds__(256) void conv_kernel(const __bf16* __restrict__ U,
                                                   const __bf16* __restrict__ Wb,
                                                   const float* __restrict__ bias,
                                                   float* __restrict__ out) {
    __shared__ float lds[8][16 * 32];          // 8 waves * 2KB
    const int lane = threadIdx.x & 31;
    const int wv   = threadIdx.x >> 5;
    const int t    = blockIdx.x * 8 + wv;      // tile id, 4*256*16 total
    const int b    = t >> 12;
    const int rr   = t & 4095;
    const int h    = rr >> 4;
    const int w0   = (rr & 15) << 4;

    const int m  = lane & 15;                  // A-matrix row (pixel in tile)
    const int kc = (lane >> 4) << 3;           // first K chunk base: 0 or 8

    f32x8 acc0 = {}, acc1 = {};

    const __bf16* urow = U + (((size_t)b * HP + h) * WP + w0 + m) * CIN + kc;

    #pragma unroll
    for (int ki = 0; ki < KH; ++ki) {
        #pragma unroll
        for (int kj = 0; kj < KW; ++kj) {
            AFrag a;
            const __bf16* ap = urow + ((size_t)ki * WP + kj) * CIN;
            a.h[0] = *(const bf16x8*)(ap);          // channels kc .. kc+7
            a.h[1] = *(const bf16x8*)(ap + 16);     // channels kc+16 .. kc+23

            const __bf16* bp = Wb + (ki * KW + kj) * 1024 + lane * 16;
            AFrag bf0, bf1;
            bf0.h[0] = *(const bf16x8*)(bp);
            bf0.h[1] = *(const bf16x8*)(bp + 8);
            bf1.h[0] = *(const bf16x8*)(bp + 512);
            bf1.h[1] = *(const bf16x8*)(bp + 520);

            acc0 = __builtin_amdgcn_wmma_f32_16x16x32_bf16(
                       false, a.v, false, bf0.v, (short)0, acc0, false, false);
            acc1 = __builtin_amdgcn_wmma_f32_16x16x32_bf16(
                       false, a.v, false, bf1.v, (short)0, acc1, false, false);
        }
    }

    // bias + relu, stage NHWC tile in LDS
    float bv0 = bias[m];
    float bv1 = bias[16 + m];
    float* tl = lds[wv];
    #pragma unroll
    for (int r = 0; r < 8; ++r) {
        int M = r + ((lane >> 4) << 3);        // pixel index in tile
        tl[M * 32 + m]      = fmaxf(acc0[r] + bv0, 0.f);
        tl[M * 32 + 16 + m] = fmaxf(acc1[r] + bv1, 0.f);
    }
    __syncthreads();

    // exp0: lane pair per pixel; ds_swizzle SWAPX1 combines the two halves
    int p  = lane >> 1;
    int ch = (lane & 1) << 4;
    const float* src = tl + p * 32 + ch;
    float ss = 0.f;
    #pragma unroll
    for (int j = 0; j < 16; ++j) { float v = src[j]; ss += v * v; }
    ss += __int_as_float(__builtin_amdgcn_ds_swizzle(__float_as_int(ss), 0x041F));

    float n  = fmaxf(sqrtf(ss), EPSN);
    float tt = SQC * n;
    float f  = tanhf(tt) / tt;

    float* op = out + (((size_t)b * COUT + ch) * HH + h) * WW + (w0 + p);
    #pragma unroll
    for (int j = 0; j < 16; ++j)
        op[(size_t)j * HH * WW] = f * src[j];
}

// ---------------------------------------------------------------------------
extern "C" void kernel_launch(void* const* d_in, const int* in_sizes, int n_in,
                              void* d_out, int out_size, void* d_ws, size_t ws_size,
                              hipStream_t stream) {
    const float* x    = (const float*)d_in[0];   // [4,32,256,256]
    const float* Wk   = (const float*)d_in[1];   // [9,32,32]
    const float* bias = (const float*)d_in[2];   // [32]
    float* out = (float*)d_out;

    __bf16* U  = (__bf16*)d_ws;                               // 4*258*258*32 bf16
    __bf16* Wb = (__bf16*)((char*)d_ws +
                 (size_t)BB * HP * WP * CIN * sizeof(__bf16)); // 9216 bf16

    int npix = BB * HP * WP;
    log0_kernel<<<(npix + 255) / 256, 256, 0, stream>>>(x, U);
    wswz_kernel<<<(9216 + 255) / 256, 256, 0, stream>>>(Wk, Wb);

    int ntiles = BB * HH * (WW / 16);            // 16384 waves
    conv_kernel<<<ntiles / 8, 256, 0, stream>>>(U, Wb, bias, out);
}
